// GATLayer_56341380989573
// MI455X (gfx1250) — matrix operand order repaired and checked by reading, not compiled, not run on hardware
//
#include <hip/hip_runtime.h>
#include <hip/hip_bf16.h>

typedef __attribute__((ext_vector_type(16))) _Float16 v16h;
typedef __attribute__((ext_vector_type(8)))  float    v8f;

#define DIM   128
#define HEADS 8
#define PHC   16

// ---------- order-preserving float<->uint encoding for atomicMax ----------
__device__ __forceinline__ unsigned enc_f32(float f) {
  unsigned b = __float_as_uint(f);
  return (b & 0x80000000u) ? ~b : (b | 0x80000000u);
}
__device__ __forceinline__ float dec_f32(unsigned e) {
  unsigned b = (e & 0x80000000u) ? (e & 0x7FFFFFFFu) : ~e;
  return __uint_as_float(b);
}

// ---------- init: zero d_out, denom; segmax <- 0 (== -max encoded) ----------
__global__ void gat_init_kernel(float* __restrict__ out, float* __restrict__ denom,
                                unsigned* __restrict__ segmax, int N) {
  int i = blockIdx.x * blockDim.x + threadIdx.x;
  if (i < N * DIM) out[i] = 0.0f;
  if (i < N * HEADS) { denom[i] = 0.0f; segmax[i] = 0u; }
}

// ---------- pack W (f32 [K=128][N=128]) into WMMA B-fragment lane layout ----
// B 32x16 f16 fragment: lane 0-15 col N=lane, elems e=K0..7,K16..23;
// lane 16-31 col N=lane-16, elems e=K8..15,K24..31.  Layout: [kk][tile][lane][e]
__global__ void gat_pack_w_kernel(const float* __restrict__ W, _Float16* __restrict__ Wpack) {
  int idx  = blockIdx.x * blockDim.x + threadIdx.x;   // 4*8*32*16 = 16384
  int e    = idx & 15;
  int lane = (idx >> 4) & 31;
  int t    = (idx >> 9) & 7;
  int kk   = idx >> 12;
  int K    = kk * 32 + (e & 7) + ((lane >> 4) << 3) + ((e >> 3) << 4);
  int col  = t * 16 + (lane & 15);
  Wpack[idx] = (_Float16)W[K * DIM + col];
}

// ---------- GEMM: nt = nodes @ W + b, one wave -> 16 rows x 128 cols --------
__global__ void __launch_bounds__(128)
gat_gemm_kernel(const float* __restrict__ nodes, const _Float16* __restrict__ Wpack,
                const float* __restrict__ bias, float* __restrict__ nt, int N) {
  const int wave = threadIdx.x >> 5;           // 0..3
  const int lane = threadIdx.x & 31;
  const int m0   = (blockIdx.x * 4 + wave) * 16;
  if (m0 >= N) return;                          // wave-uniform: EXEC stays full
  const int c  = lane & 15;
  const int hi = lane >> 4;
  const int mrow = m0 + c;
  const float* rowp = nodes + (size_t)(mrow < N ? mrow : N - 1) * DIM;

  v8f acc[8] = {};
#pragma unroll
  for (int kk = 0; kk < 4; ++kk) {
    // A 16x32 f16 fragment (ISA 7.12.2): lane-lo holds K=kb..kb+7,kb+16..23 of
    // row M=lane; lane-hi holds K offset by +8 of row M=lane-16.
    const int kb = kk * 32 + hi * 8;
    const float4* p0 = reinterpret_cast<const float4*>(rowp + kb);
    const float4* p1 = reinterpret_cast<const float4*>(rowp + kb + 16);
    float4 q0 = p0[0], q1 = p0[1], q2 = p1[0], q3 = p1[1];
    v16h a;
    a[0]  = (_Float16)q0.x; a[1]  = (_Float16)q0.y; a[2]  = (_Float16)q0.z; a[3]  = (_Float16)q0.w;
    a[4]  = (_Float16)q1.x; a[5]  = (_Float16)q1.y; a[6]  = (_Float16)q1.z; a[7]  = (_Float16)q1.w;
    a[8]  = (_Float16)q2.x; a[9]  = (_Float16)q2.y; a[10] = (_Float16)q2.z; a[11] = (_Float16)q2.w;
    a[12] = (_Float16)q3.x; a[13] = (_Float16)q3.y; a[14] = (_Float16)q3.z; a[15] = (_Float16)q3.w;
#pragma unroll
    for (int t = 0; t < 8; ++t) {
      v16h b = *reinterpret_cast<const v16h*>(Wpack + ((size_t)(kk * 8 + t) * 32 + lane) * 16);
      acc[t] = __builtin_amdgcn_wmma_f32_16x16x32_f16(
          /*neg_a=*/false, a, /*neg_b=*/false, b,
          /*c_mod=*/(short)0, acc[t], /*reuse_a=*/false, /*reuse_b=*/false);
    }
  }
  // D layout: VGPR v -> row m0 + v + hi*8, col t*16 + (lane&15)
#pragma unroll
  for (int t = 0; t < 8; ++t) {
    const int col = t * 16 + c;
    const float bv = bias[col];
#pragma unroll
    for (int v = 0; v < 8; ++v) {
      const int m = m0 + v + hi * 8;
      if (m < N) nt[(size_t)m * DIM + col] = acc[t][v] + bv;
    }
  }
}

// ---------- per-node separable attention coefficients ----------------------
// a_s[n,h] = sum_c lrelu(nt[n,h,c]) * aw[c];  a_r uses aw[16+c]
__global__ void gat_coef_kernel(const float* __restrict__ nt, const float* __restrict__ aw,
                                float* __restrict__ a_s, float* __restrict__ a_r, int NH) {
  int idx = blockIdx.x * blockDim.x + threadIdx.x;   // n*HEADS + h
  if (idx >= NH) return;
  const float* p = nt + (size_t)idx * PHC;
  float s = 0.f, r = 0.f;
#pragma unroll
  for (int c = 0; c < PHC; ++c) {
    float v  = p[c];
    float lr = v > 0.f ? v : 0.2f * v;
    s += lr * aw[c];
    r += lr * aw[PHC + c];
  }
  a_s[idx] = s; a_r[idx] = r;
}

// ---------- pass 1: segment max over receivers -----------------------------
__global__ void gat_edge_max_kernel(const int* __restrict__ snd, const int* __restrict__ rcv,
                                    const float* __restrict__ a_s, const float* __restrict__ a_r,
                                    const float* __restrict__ attn_b,
                                    unsigned* __restrict__ segmax, int E) {
  int e = blockIdx.x * blockDim.x + threadIdx.x;
  if (e >= E) return;
  int s = snd[e], r = rcv[e];
  const float4* ps = reinterpret_cast<const float4*>(a_s + (size_t)s * HEADS);
  const float4* pr = reinterpret_cast<const float4*>(a_r + (size_t)r * HEADS);
  float4 s0 = ps[0], s1 = ps[1], r0 = pr[0], r1 = pr[1];
  float ab = attn_b[0];
  float lg[8] = { s0.x + r0.x + ab, s0.y + r0.y + ab, s0.z + r0.z + ab, s0.w + r0.w + ab,
                  s1.x + r1.x + ab, s1.y + r1.y + ab, s1.z + r1.z + ab, s1.w + r1.w + ab };
#pragma unroll
  for (int h = 0; h < HEADS; ++h)
    atomicMax(&segmax[(size_t)r * HEADS + h], enc_f32(lg[h]));
}

// ---------- pass 2: softmax denominator ------------------------------------
__global__ void gat_edge_sum_kernel(const int* __restrict__ snd, const int* __restrict__ rcv,
                                    const float* __restrict__ a_s, const float* __restrict__ a_r,
                                    const float* __restrict__ attn_b,
                                    const unsigned* __restrict__ segmax,
                                    float* __restrict__ denom, int E) {
  int e = blockIdx.x * blockDim.x + threadIdx.x;
  if (e >= E) return;
  int s = snd[e], r = rcv[e];
  const float4* ps = reinterpret_cast<const float4*>(a_s + (size_t)s * HEADS);
  const float4* pr = reinterpret_cast<const float4*>(a_r + (size_t)r * HEADS);
  float4 s0 = ps[0], s1 = ps[1], r0 = pr[0], r1 = pr[1];
  float ab = attn_b[0];
  float lg[8] = { s0.x + r0.x + ab, s0.y + r0.y + ab, s0.z + r0.z + ab, s0.w + r0.w + ab,
                  s1.x + r1.x + ab, s1.y + r1.y + ab, s1.z + r1.z + ab, s1.w + r1.w + ab };
#pragma unroll
  for (int h = 0; h < HEADS; ++h) {
    float m = dec_f32(segmax[(size_t)r * HEADS + h]);
    atomicAdd(&denom[(size_t)r * HEADS + h], __expf(lg[h] - m));
  }
}

// ---------- pass 3: alpha-weighted scatter-sum; 128 threads per edge -------
__global__ void gat_edge_scatter_kernel(const int* __restrict__ snd, const int* __restrict__ rcv,
                                        const float* __restrict__ a_s, const float* __restrict__ a_r,
                                        const float* __restrict__ attn_b,
                                        const unsigned* __restrict__ segmax,
                                        const float* __restrict__ denom,
                                        const float* __restrict__ nt,
                                        float* __restrict__ out, int E) {
  unsigned gid = blockIdx.x * blockDim.x + threadIdx.x;
  int e = (int)(gid >> 7);
  if (e >= E) return;
  int l = (int)(gid & 127u);
  int h = l >> 4, c = l & 15;
  int s = snd[e], r = rcv[e];
  float logit = a_s[(size_t)s * HEADS + h] + a_r[(size_t)r * HEADS + h] + attn_b[0];
  float m     = dec_f32(segmax[(size_t)r * HEADS + h]);
  float alpha = __expf(logit - m) / denom[(size_t)r * HEADS + h];
  float v     = nt[(size_t)s * DIM + h * PHC + c] * alpha;
  atomicAdd(&out[(size_t)r * DIM + h * PHC + c], v);
}

extern "C" void kernel_launch(void* const* d_in, const int* in_sizes, int n_in,
                              void* d_out, int out_size, void* d_ws, size_t ws_size,
                              hipStream_t stream) {
  const float* nodes  = (const float*)d_in[0];
  const int*   snd    = (const int*)  d_in[1];
  const int*   rcv    = (const int*)  d_in[2];
  const float* W      = (const float*)d_in[3];
  const float* bias   = (const float*)d_in[4];
  const float* attn_w = (const float*)d_in[5];
  const float* attn_b = (const float*)d_in[6];
  float*       out    = (float*)d_out;

  const int N = in_sizes[0] / DIM;
  const int E = in_sizes[1];

  // workspace carve-up (~64 MB): nt | a_s | a_r | segmax | denom | Wpack
  char* ws = (char*)d_ws;
  float*     nt     = (float*)ws;                 ws += (size_t)N * DIM   * sizeof(float);
  float*     a_s    = (float*)ws;                 ws += (size_t)N * HEADS * sizeof(float);
  float*     a_r    = (float*)ws;                 ws += (size_t)N * HEADS * sizeof(float);
  unsigned*  segmax = (unsigned*)ws;              ws += (size_t)N * HEADS * sizeof(unsigned);
  float*     denom  = (float*)ws;                 ws += (size_t)N * HEADS * sizeof(float);
  _Float16*  Wpack  = (_Float16*)ws;              ws += (size_t)4 * 8 * 32 * 16 * sizeof(_Float16);
  (void)ws_size; (void)n_in; (void)out_size;

  const int blk = 256;

  gat_init_kernel<<<(N * DIM + blk - 1) / blk, blk, 0, stream>>>(out, denom, segmax, N);

  gat_pack_w_kernel<<<(4 * 8 * 32 * 16) / blk, blk, 0, stream>>>(W, Wpack);

  // 128 threads = 4 waves; each wave does a 16x128 strip via 32 WMMAs
  gat_gemm_kernel<<<(N + 63) / 64, 128, 0, stream>>>(nodes, Wpack, bias, nt, N);

  gat_coef_kernel<<<(N * HEADS + blk - 1) / blk, blk, 0, stream>>>(nt, attn_w, a_s, a_r, N * HEADS);

  gat_edge_max_kernel<<<(E + blk - 1) / blk, blk, 0, stream>>>(snd, rcv, a_s, a_r, attn_b, segmax, E);

  gat_edge_sum_kernel<<<(E + blk - 1) / blk, blk, 0, stream>>>(snd, rcv, a_s, a_r, attn_b, segmax, denom, E);

  unsigned long long sc_threads = (unsigned long long)E * DIM;
  gat_edge_scatter_kernel<<<(unsigned)((sc_threads + blk - 1) / blk), blk, 0, stream>>>(
      snd, rcv, a_s, a_r, attn_b, segmax, denom, nt, out, E);
}
